// CROMA_JOINT_Encoder_13743895347306
// MI455X (gfx1250) — matrix operand order repaired and checked by reading, not compiled, not run on hardware
//
#include <hip/hip_runtime.h>
#include <hip/hip_bf16.h>
#include <math.h>

// ---------------------------------------------------------------------------
// CROMA joint encoder forward for gfx1250 (MI455X).
// All matmuls go through v_wmma_f32_16x16x32_f16 (wave32 WMMA), f32 accum.
// ---------------------------------------------------------------------------

#define DIMX   768
#define HEADS  16
#define NP     225
#define NPAD   256
#define DH     48
#define DHPAD  64
#define BATCHX 16
#define ROWS   (BATCHX * NP)     // 3600
#define BH     (BATCHX * HEADS)  // 256

typedef _Float16 h4   __attribute__((ext_vector_type(4)));
typedef _Float16 h8   __attribute__((ext_vector_type(8)));
typedef _Float16 v16h __attribute__((ext_vector_type(16)));
typedef float    v8f  __attribute__((ext_vector_type(8)));
typedef float    f4   __attribute__((ext_vector_type(4)));

// ---------------------------------------------------------------------------
// Generic tiled WMMA GEMM:  C[M,N] = A[M,K] @ B[K,N]  (+ epilogue)
// A, B are f32 row-major in global memory, converted to f16 while staged into
// LDS (double buffered).  8 waves per block, each wave computes a 32x64 tile
// via 2x4 v_wmma_f32_16x16x32_f16.
// EPI: 0 none, 1 +bias, 2 +bias+residual, 3 gelu(x+bias), 4 +alibi(h,i,j)
// ---------------------------------------------------------------------------
template<int BM, int BN, int WM, int WN, int EPI>
__global__ __launch_bounds__(256) void gemm_wmma_k(
    const float* __restrict__ A, int lda, long long sA,
    const float* __restrict__ Bm, int ldb, long long sB,
    float* __restrict__ C, int ldc, long long sC,
    const float* __restrict__ bias, const float* __restrict__ res,
    int M, int N, int K)
{
    constexpr int BK  = 32;
    constexpr int AS  = BK + 8;   // LDS A row stride (halves), 16B-aligned rows
    constexpr int BS  = BN + 8;   // LDS B row stride (halves), multiple of 8
    constexpr int NT  = 256;
    constexpr int ALD = (BM * BK) / (NT * 4);
    constexpr int BLD = (BK * BN) / (NT * 4);

    __shared__ _Float16 Asm[2][BM * AS];
    __shared__ _Float16 Bsm[2][BK * BS];

    const int tid  = threadIdx.x;
    const int lane = tid & 31;
    const int wid  = tid >> 5;
    const int wm   = wid % WM;
    const int wn   = wid / WM;
    const int m0   = blockIdx.y * BM;
    const int n0   = blockIdx.x * BN;

    const float* Ab = A  + (long long)blockIdx.z * sA;
    const float* Bb = Bm + (long long)blockIdx.z * sB;

    f4 ra[ALD];
    f4 rb[BLD];

    auto fetch = [&](int ks) {
        const int k0 = ks * BK;
#pragma unroll
        for (int i = 0; i < ALD; ++i) {
            int e = (tid + i * NT) * 4;
            int r = e / BK, c = e % BK;
            int m = m0 + r;
            if (m < M) {
                ra[i] = *(const f4*)(Ab + (long long)m * lda + (k0 + c));
            } else {
                ra[i][0] = 0.f; ra[i][1] = 0.f; ra[i][2] = 0.f; ra[i][3] = 0.f;
            }
        }
#pragma unroll
        for (int i = 0; i < BLD; ++i) {
            int e = (tid + i * NT) * 4;
            int r = e / BN, c = e % BN;
            rb[i] = *(const f4*)(Bb + (long long)(k0 + r) * ldb + (n0 + c));
        }
    };

    auto stage = [&](int buf) {
#pragma unroll
        for (int i = 0; i < ALD; ++i) {
            int e = (tid + i * NT) * 4;
            int r = e / BK, c = e % BK;
            h4 hv;
            hv[0] = (_Float16)ra[i][0]; hv[1] = (_Float16)ra[i][1];
            hv[2] = (_Float16)ra[i][2]; hv[3] = (_Float16)ra[i][3];
            *(h4*)(&Asm[buf][r * AS + c]) = hv;
        }
#pragma unroll
        for (int i = 0; i < BLD; ++i) {
            int e = (tid + i * NT) * 4;
            int r = e / BN, c = e % BN;
            h4 hv;
            hv[0] = (_Float16)rb[i][0]; hv[1] = (_Float16)rb[i][1];
            hv[2] = (_Float16)rb[i][2]; hv[3] = (_Float16)rb[i][3];
            *(h4*)(&Bsm[buf][r * BS + c]) = hv;
        }
    };

    v8f acc[2][4];
#pragma unroll
    for (int mi = 0; mi < 2; ++mi)
#pragma unroll
        for (int ni = 0; ni < 4; ++ni)
#pragma unroll
            for (int r = 0; r < 8; ++r) acc[mi][ni][r] = 0.f;

    auto compute = [&](int buf) {
        const _Float16* As = &Asm[buf][0];
        const _Float16* Bs = &Bsm[buf][0];
        // A fragment (ISA 16-bit A 16x32): lane -> M=lane&15,
        // K halves [kb..kb+7] and [kb+16..kb+23], kb = (lane>>4)*8
        const int am = wm * 32 + (lane & 15);
        const int kb = (lane >> 4) * 8;
        v16h af[2];
#pragma unroll
        for (int mi = 0; mi < 2; ++mi) {
            const _Float16* ap = As + (am + mi * 16) * AS;
            h8 lo = *(const h8*)(ap + kb);
            h8 hi = *(const h8*)(ap + kb + 16);
#pragma unroll
            for (int q = 0; q < 8; ++q) { af[mi][q] = lo[q]; af[mi][q + 8] = hi[q]; }
        }
        // B fragment: lane -> N=lane&15, K rows [kb..kb+7] and [kb+16..kb+23]
#pragma unroll
        for (int ni = 0; ni < 4; ++ni) {
            const _Float16* bp = Bs + wn * 64 + ni * 16 + (lane & 15);
            v16h bf;
#pragma unroll
            for (int q = 0; q < 8; ++q) { bf[q] = bp[(kb + q) * BS]; bf[q + 8] = bp[(kb + 16 + q) * BS]; }
#pragma unroll
            for (int mi = 0; mi < 2; ++mi)
                acc[mi][ni] = __builtin_amdgcn_wmma_f32_16x16x32_f16(
                    false, af[mi], false, bf, (short)0, acc[mi][ni], false, false);
        }
    };

    const int nK = K / BK;
    fetch(0); stage(0);
    __syncthreads();
    for (int ks = 0; ks < nK; ++ks) {
        const int curb = ks & 1;
        if (ks + 1 < nK) fetch(ks + 1);
        compute(curb);
        __syncthreads();
        if (ks + 1 < nK) { stage(curb ^ 1); __syncthreads(); }
    }

    // Epilogue.  C/D layout: m_local = 8*(lane>>4) + vgpr, n_local = lane&15.
    float* Cb = C + (long long)blockIdx.z * sC;
#pragma unroll
    for (int mi = 0; mi < 2; ++mi) {
#pragma unroll
        for (int ni = 0; ni < 4; ++ni) {
#pragma unroll
            for (int r = 0; r < 8; ++r) {
                int m = m0 + wm * 32 + mi * 16 + (lane >> 4) * 8 + r;
                int n = n0 + wn * 64 + ni * 16 + (lane & 15);
                if (m < M && n < N) {
                    float v = acc[mi][ni][r];
                    if (EPI == 1) {
                        v += bias[n];
                    } else if (EPI == 2) {
                        v += bias[n] + res[(long long)m * ldc + n];
                    } else if (EPI == 3) {
                        v += bias[n];
                        v = 0.5f * v * (1.0f + erff(v * 0.70710678118654752f));
                    } else if (EPI == 4) {
                        if (m < NP && n < NP) {
                            int h = blockIdx.z & (HEADS - 1);
                            float slope = exp2f(-0.5f * (float)(h + 1));
                            float dx = (float)(m / 15 - n / 15);
                            float dy = (float)(m % 15 - n % 15);
                            v += -slope * sqrtf(dx * dx + dy * dy);
                        }
                    }
                    Cb[(long long)m * ldc + n] = v;
                }
            }
        }
    }
}

// ---------------------------------------------------------------------------
// Elementwise / reduction kernels
// ---------------------------------------------------------------------------
__global__ __launch_bounds__(256) void patchify_k(
    const float* __restrict__ img, float* __restrict__ out, int Cch, long long total)
{
    long long idx = (long long)blockIdx.x * 256 + threadIdx.x;
    if (idx >= total) return;
    int Kc = Cch * 64;
    int col = (int)(idx % Kc);
    long long row = idx / Kc;
    int b = (int)(row / NP), p = (int)(row % NP);
    int gi = p / 15, gj = p % 15;
    int c = col / 64, r8 = (col % 64) / 8, c8 = col % 8;
    out[idx] = img[(((long long)b * Cch + c) * 120 + gi * 8 + r8) * 120 + gj * 8 + c8];
}

__global__ __launch_bounds__(256) void layernorm_k(
    const float* __restrict__ x, const float* __restrict__ g,
    const float* __restrict__ b, float* __restrict__ y)
{
    __shared__ float red[256];
    int tid = threadIdx.x;
    long long base = (long long)blockIdx.x * DIMX;
    float v0 = x[base + tid], v1 = x[base + tid + 256], v2 = x[base + tid + 512];
    red[tid] = v0 + v1 + v2;
    __syncthreads();
    for (int o = 128; o > 0; o >>= 1) { if (tid < o) red[tid] += red[tid + o]; __syncthreads(); }
    float mean = red[0] * (1.f / 768.f);
    __syncthreads();
    float d0 = v0 - mean, d1 = v1 - mean, d2 = v2 - mean;
    red[tid] = d0 * d0 + d1 * d1 + d2 * d2;
    __syncthreads();
    for (int o = 128; o > 0; o >>= 1) { if (tid < o) red[tid] += red[tid + o]; __syncthreads(); }
    float rstd = rsqrtf(red[0] * (1.f / 768.f) + 1e-5f);
    y[base + tid]       = d0 * rstd * g[tid]       + b[tid];
    y[base + tid + 256] = d1 * rstd * g[tid + 256] + b[tid + 256];
    y[base + tid + 512] = d2 * rstd * g[tid + 512] + b[tid + 512];
}

// Split heads into zero-padded [bh][256][64] q/v and transposed [bh][64][256] k.
__global__ __launch_bounds__(256) void split_heads_k(
    const float* __restrict__ q, const float* __restrict__ k, const float* __restrict__ v,
    int rstride, float* __restrict__ qh, float* __restrict__ kT, float* __restrict__ vh,
    float scale)
{
    int idx = blockIdx.x * 256 + threadIdx.x;   // BH*256*64 total
    int d  = idx & 63;
    int n  = (idx >> 6) & 255;
    int bh = idx >> 14;
    int b = bh >> 4, h = bh & 15;
    float qv = 0.f, kv = 0.f, vv = 0.f;
    if (n < NP && d < DH) {
        long long s = (long long)(b * NP + n) * rstride + h * DH + d;
        qv = q[s] * scale; kv = k[s]; vv = v[s];
    }
    qh[idx] = qv;
    vh[idx] = vv;
    kT[((long long)bh << 14) + d * NPAD + n] = kv;
}

__global__ __launch_bounds__(256) void softmax_k(float* __restrict__ S)
{
    __shared__ float red[256];
    int bh = blockIdx.x / NP;
    int i  = blockIdx.x % NP;
    long long base = ((long long)bh * NPAD + i) * NPAD;
    int tid = threadIdx.x;
    float x = (tid < NP) ? S[base + tid] : -3.0e38f;
    red[tid] = x;
    __syncthreads();
    for (int o = 128; o > 0; o >>= 1) { if (tid < o) red[tid] = fmaxf(red[tid], red[tid + o]); __syncthreads(); }
    float mx = red[0];
    __syncthreads();
    float e = (tid < NP) ? expf(x - mx) : 0.f;
    red[tid] = e;
    __syncthreads();
    for (int o = 128; o > 0; o >>= 1) { if (tid < o) red[tid] += red[tid + o]; __syncthreads(); }
    float inv = 1.f / red[0];
    S[base + tid] = e * inv;   // padded columns become exact zeros
}

__global__ __launch_bounds__(256) void merge_heads_k(
    const float* __restrict__ O, float* __restrict__ xo)
{
    int idx = blockIdx.x * 256 + threadIdx.x;   // ROWS*DIMX total
    int dd = idx % DIMX;
    int rn = idx / DIMX;
    int b = rn / NP, n = rn % NP;
    int h = dd / DH, d = dd % DH;
    xo[idx] = O[(((long long)(b * HEADS + h)) * NPAD + n) * DHPAD + d];
}

// x[b*225+p][d] -> out[b][d][p], with optional final LayerNorm (g != nullptr).
__global__ __launch_bounds__(256) void out_tr_k(
    const float* __restrict__ x, const float* __restrict__ g,
    const float* __restrict__ bb, float* __restrict__ out)
{
    __shared__ float red[256];
    int tid = threadIdx.x;
    int bp = blockIdx.x;
    int b = bp / NP, p = bp % NP;
    long long base = (long long)bp * DIMX;
    float v0 = x[base + tid], v1 = x[base + tid + 256], v2 = x[base + tid + 512];
    if (g) {
        red[tid] = v0 + v1 + v2;
        __syncthreads();
        for (int o = 128; o > 0; o >>= 1) { if (tid < o) red[tid] += red[tid + o]; __syncthreads(); }
        float mean = red[0] * (1.f / 768.f);
        __syncthreads();
        float d0 = v0 - mean, d1 = v1 - mean, d2 = v2 - mean;
        red[tid] = d0 * d0 + d1 * d1 + d2 * d2;
        __syncthreads();
        for (int o = 128; o > 0; o >>= 1) { if (tid < o) red[tid] += red[tid + o]; __syncthreads(); }
        float rstd = rsqrtf(red[0] * (1.f / 768.f) + 1e-5f);
        v0 = d0 * rstd * g[tid]       + bb[tid];
        v1 = d1 * rstd * g[tid + 256] + bb[tid + 256];
        v2 = d2 * rstd * g[tid + 512] + bb[tid + 512];
    }
    float* ob = out + (long long)b * DIMX * NP;
    ob[(long long)tid * NP + p]         = v0;
    ob[(long long)(tid + 256) * NP + p] = v1;
    ob[(long long)(tid + 512) * NP + p] = v2;
}

// ---------------------------------------------------------------------------
// Host-side orchestration
// ---------------------------------------------------------------------------
struct AttnP  { const float *ln_g, *ln_b, *w_qkv, *w_out, *b_out; };
struct XAttnP { const float *ln_g, *ln_b, *w_q, *w_k, *w_v, *w_out, *b_out; };
struct FfnP   { const float *ln_g, *ln_b, *w1, *b1, *w2, *b2; };
struct VitP   { const float *in_w, *in_b, *norm_g, *norm_b; AttnP attn[12]; FfnP ffn[12]; int depth; };
struct CrossP { AttnP attn[6]; XAttnP xattn[6]; FfnP ffn[6]; const float *norm_g, *norm_b; };

struct Work { float *xn, *cn, *big, *qh, *kT, *vh, *S, *O; };

static void gemm128(hipStream_t st, int epi,
                    const float* A, int lda, long long sA,
                    const float* B, int ldb, long long sB,
                    float* C, int ldc, long long sC,
                    const float* bias, const float* res,
                    int M, int N, int K, int batches)
{
    dim3 g(N / 128, (M + 127) / 128, batches), blk(256, 1, 1);
    switch (epi) {
    case 0: gemm_wmma_k<128,128,4,2,0><<<g,blk,0,st>>>(A,lda,sA,B,ldb,sB,C,ldc,sC,bias,res,M,N,K); break;
    case 1: gemm_wmma_k<128,128,4,2,1><<<g,blk,0,st>>>(A,lda,sA,B,ldb,sB,C,ldc,sC,bias,res,M,N,K); break;
    case 2: gemm_wmma_k<128,128,4,2,2><<<g,blk,0,st>>>(A,lda,sA,B,ldb,sB,C,ldc,sC,bias,res,M,N,K); break;
    case 3: gemm_wmma_k<128,128,4,2,3><<<g,blk,0,st>>>(A,lda,sA,B,ldb,sB,C,ldc,sC,bias,res,M,N,K); break;
    case 4: gemm_wmma_k<128,128,4,2,4><<<g,blk,0,st>>>(A,lda,sA,B,ldb,sB,C,ldc,sC,bias,res,M,N,K); break;
    }
}

static void gemmAV(hipStream_t st,
                   const float* A, int lda, long long sA,
                   const float* B, int ldb, long long sB,
                   float* C, int ldc, long long sC,
                   int M, int N, int K, int batches)
{
    dim3 g(N / 64, (M + 255) / 256, batches), blk(256, 1, 1);
    gemm_wmma_k<256,64,8,1,0><<<g,blk,0,st>>>(A,lda,sA,B,ldb,sB,C,ldc,sC,nullptr,nullptr,M,N,K);
}

static void attn_tail(hipStream_t st, Work& W, float* x,
                      const float* q, const float* k, const float* v, int rstride,
                      const float* w_out, const float* b_out)
{
    split_heads_k<<<(BH * NPAD * DHPAD) / 256, 256, 0, st>>>(
        q, k, v, rstride, W.qh, W.kT, W.vh, 0.14433756729740643f /* 48^-0.5 */);
    // scores: [bh] Q[256,64] @ Kt[64,256] + alibi
    gemm128(st, 4, W.qh, DHPAD, (long long)NPAD * DHPAD,
            W.kT, NPAD, (long long)NPAD * DHPAD,
            W.S, NPAD, (long long)NPAD * NPAD,
            nullptr, nullptr, NPAD, NPAD, DHPAD, BH);
    softmax_k<<<BH * NP, 256, 0, st>>>(W.S);
    // O = P @ V : [bh] [256,256] @ [256,64]
    gemmAV(st, W.S, NPAD, (long long)NPAD * NPAD,
           W.vh, DHPAD, (long long)NPAD * DHPAD,
           W.O, DHPAD, (long long)NPAD * DHPAD,
           NPAD, DHPAD, NPAD, BH);
    merge_heads_k<<<(ROWS * DIMX) / 256, 256, 0, st>>>(W.O, W.xn);  // reuse xn as merged
    gemm128(st, 2, W.xn, DIMX, 0, w_out, DIMX, 0, x, DIMX, 0,
            b_out, x, ROWS, DIMX, DIMX, 1);
}

static void self_attn(hipStream_t st, Work& W, float* x, const AttnP& p)
{
    layernorm_k<<<ROWS, 256, 0, st>>>(x, p.ln_g, p.ln_b, W.xn);
    gemm128(st, 0, W.xn, DIMX, 0, p.w_qkv, 3 * DIMX, 0, W.big, 3 * DIMX, 0,
            nullptr, nullptr, ROWS, 3 * DIMX, DIMX, 1);
    attn_tail(st, W, x, W.big, W.big + DIMX, W.big + 2 * DIMX, 3 * DIMX, p.w_out, p.b_out);
}

static void cross_attn(hipStream_t st, Work& W, float* x, const float* ctx, const XAttnP& p)
{
    layernorm_k<<<ROWS, 256, 0, st>>>(x,   p.ln_g, p.ln_b, W.xn);
    layernorm_k<<<ROWS, 256, 0, st>>>(ctx, p.ln_g, p.ln_b, W.cn);
    float* qb = W.big;
    float* kb = W.big + (size_t)ROWS * DIMX;
    float* vb = kb + (size_t)ROWS * DIMX;
    gemm128(st, 0, W.xn, DIMX, 0, p.w_q, DIMX, 0, qb, DIMX, 0, nullptr, nullptr, ROWS, DIMX, DIMX, 1);
    gemm128(st, 0, W.cn, DIMX, 0, p.w_k, DIMX, 0, kb, DIMX, 0, nullptr, nullptr, ROWS, DIMX, DIMX, 1);
    gemm128(st, 0, W.cn, DIMX, 0, p.w_v, DIMX, 0, vb, DIMX, 0, nullptr, nullptr, ROWS, DIMX, DIMX, 1);
    attn_tail(st, W, x, qb, kb, vb, DIMX, p.w_out, p.b_out);
}

static void ffn_blk(hipStream_t st, Work& W, float* x, const FfnP& p)
{
    layernorm_k<<<ROWS, 256, 0, st>>>(x, p.ln_g, p.ln_b, W.xn);
    gemm128(st, 3, W.xn, DIMX, 0, p.w1, 4 * DIMX, 0, W.big, 4 * DIMX, 0,
            p.b1, nullptr, ROWS, 4 * DIMX, DIMX, 1);
    gemm128(st, 2, W.big, 4 * DIMX, 0, p.w2, DIMX, 0, x, DIMX, 0,
            p.b2, x, ROWS, DIMX, 4 * DIMX, 1);
}

static void vit_run(hipStream_t st, Work& W, const float* img, int Cch,
                    const VitP& p, float* x, float* enc)
{
    long long tot = (long long)ROWS * Cch * 64;
    patchify_k<<<(unsigned)((tot + 255) / 256), 256, 0, st>>>(img, W.xn, Cch, tot);
    gemm128(st, 1, W.xn, Cch * 64, 0, p.in_w, DIMX, 0, x, DIMX, 0,
            p.in_b, nullptr, ROWS, DIMX, Cch * 64, 1);
    for (int i = 0; i < p.depth; ++i) {
        self_attn(st, W, x, p.attn[i]);
        ffn_blk(st, W, x, p.ffn[i]);
    }
    layernorm_k<<<ROWS, 256, 0, st>>>(x, p.norm_g, p.norm_b, enc);
}

extern "C" void kernel_launch(void* const* d_in, const int* in_sizes, int n_in,
                              void* d_out, int out_size, void* d_ws, size_t ws_size,
                              hipStream_t stream)
{
    (void)in_sizes; (void)out_size; (void)ws_size;
    const float* sar     = (const float*)d_in[0];
    const float* optical = (const float*)d_in[1];

    // Params flattened in dict insertion order: s1, s2, cross.
    int cur = 2;
    auto nx = [&]() { const float* p = (const float*)d_in[cur < n_in ? cur : n_in - 1]; ++cur; return p; };
    auto rdAttn = [&](AttnP& a)  { a.ln_g=nx(); a.ln_b=nx(); a.w_qkv=nx(); a.w_out=nx(); a.b_out=nx(); };
    auto rdFfn  = [&](FfnP& f)   { f.ln_g=nx(); f.ln_b=nx(); f.w1=nx(); f.b1=nx(); f.w2=nx(); f.b2=nx(); };
    auto rdX    = [&](XAttnP& x) { x.ln_g=nx(); x.ln_b=nx(); x.w_q=nx(); x.w_k=nx(); x.w_v=nx(); x.w_out=nx(); x.b_out=nx(); };

    VitP s1, s2; CrossP cross;
    auto rdVit = [&](VitP& v, int depth) {
        v.in_w = nx(); v.in_b = nx();
        for (int i = 0; i < depth; ++i) { rdAttn(v.attn[i]); rdFfn(v.ffn[i]); }
        v.norm_g = nx(); v.norm_b = nx();
        v.depth = depth;
    };
    rdVit(s1, 6);
    rdVit(s2, 12);
    for (int i = 0; i < 6; ++i) { rdAttn(cross.attn[i]); rdX(cross.xattn[i]); rdFfn(cross.ffn[i]); }
    cross.norm_g = nx(); cross.norm_b = nx();

    // Workspace arena (floats).
    float* ws = (float*)d_ws;
    size_t off = 0;
    auto arena = [&](size_t n) { float* p = ws + off; off += n; return p; };
    float* bufX   = arena(2764800);     // ViT working activations
    float* bufSar = arena(2764800);     // sar encoder output -> cross working x
    float* bufCtx = arena(2764800);     // optical encoder output (context)
    Work W;
    W.xn  = arena(2764800);             // LN output / patch matrix / merged heads
    W.cn  = arena(2764800);             // context LN output
    W.big = arena(11059200);            // qkv (3600x2304) / q,k,v / ffn hidden (3600x3072)
    W.qh  = arena(4194304);             // [256][256][64]
    W.kT  = arena(4194304);             // [256][64][256]
    W.vh  = arena(4194304);             // [256][256][64]
    W.S   = arena(16777216);            // [256][256][256] scores / probs (in-place softmax)
    W.O   = arena(4194304);             // [256][256][64]

    // Two ViT encoders.
    vit_run(stream, W, sar,     2,  s1, bufX, bufSar);
    vit_run(stream, W, optical, 12, s2, bufX, bufCtx);

    // Cross encoder (x starts as sar encoding; ctx = optical encoding).
    float* x = bufSar;
    float* out = (float*)d_out;
    const long long seg = (long long)BATCHX * DIMX * NP;   // 2,764,800
    for (int i = 0; i < 6; ++i) {
        self_attn(stream, W, x, cross.attn[i]);
        cross_attn(stream, W, x, bufCtx, cross.xattn[i]);
        ffn_blk(stream, W, x, cross.ffn[i]);
        if (i == 1) out_tr_k<<<ROWS, 256, 0, stream>>>(x, nullptr, nullptr, out);
        if (i == 3) out_tr_k<<<ROWS, 256, 0, stream>>>(x, nullptr, nullptr, out + seg);
        if (i == 5) out_tr_k<<<ROWS, 256, 0, stream>>>(x, cross.norm_g, cross.norm_b, out + 2 * seg);
    }
}
